// GraphConvolutionalNetwork_84971632984099
// MI455X (gfx1250) — matrix operand order repaired and checked by reading, not compiled, not run on hardware
//
#include <hip/hip_runtime.h>

#define N_NODES 50000
#define N_EDGES 400000
#define N_GRAPHS 64
#define DK 256          // feature dim (D_IN == D_HID == 256)
#define D_OUT 16

typedef __attribute__((ext_vector_type(16))) _Float16 v16h;
typedef __attribute__((ext_vector_type(8)))  _Float16 v8h;
typedef __attribute__((ext_vector_type(8)))  float    v8f;

// ---------------------------------------------------------------- utilities
__device__ __forceinline__ void atomAddF(float* p, float v) {
    __hip_atomic_fetch_add(p, v, __ATOMIC_RELAXED, __HIP_MEMORY_SCOPE_AGENT);
}

__global__ void k_zero_i32(int* __restrict__ p, int n) {
    int i = blockIdx.x * blockDim.x + threadIdx.x;
    if (i < n) p[i] = 0;
}
__global__ void k_zero_f32(float* __restrict__ p, int n) {
    int i = blockIdx.x * blockDim.x + threadIdx.x;
    if (i < n) p[i] = 0.0f;
}

// ---------------------------------------------------------------- graph prep
__global__ void k_deg(const int* __restrict__ ei, int* __restrict__ deg) {
    int e = blockIdx.x * blockDim.x + threadIdx.x;
    if (e < N_EDGES) atomicAdd(&deg[ei[N_EDGES + e]], 1);
}

__global__ void k_dinv(const int* __restrict__ deg, float* __restrict__ dinv) {
    int i = blockIdx.x * blockDim.x + threadIdx.x;
    if (i < N_NODES) dinv[i] = rsqrtf((float)(deg[i] + 1));  // +1 self loop
}

// single-workgroup exclusive scan of deg -> rowptr, also primes cursor
__global__ __launch_bounds__(1024) void k_scan(const int* __restrict__ deg,
                                               int* __restrict__ rowptr,
                                               int* __restrict__ cursor) {
    __shared__ int sh[1024];
    const int t = threadIdx.x;
    const int CH = (N_NODES + 1023) / 1024;  // 49
    const int base = t * CH;
    int sum = 0;
    for (int j = 0; j < CH; ++j) {
        int idx = base + j;
        if (idx < N_NODES) sum += deg[idx];
    }
    sh[t] = sum;
    __syncthreads();
    for (int off = 1; off < 1024; off <<= 1) {
        int v = (t >= off) ? sh[t - off] : 0;
        __syncthreads();
        sh[t] += v;
        __syncthreads();
    }
    int run = (t == 0) ? 0 : sh[t - 1];
    for (int j = 0; j < CH; ++j) {
        int idx = base + j;
        if (idx < N_NODES) {
            rowptr[idx] = run;
            cursor[idx] = run;
            run += deg[idx];
        }
    }
    if (t == 1023) rowptr[N_NODES] = sh[1023];
}

__global__ void k_fill(const int* __restrict__ ei, int* __restrict__ cursor,
                       int* __restrict__ csr_src) {
    int e = blockIdx.x * blockDim.x + threadIdx.x;
    if (e < N_EDGES) {
        int src = ei[e];
        int dst = ei[N_EDGES + e];
        int pos = atomicAdd(&cursor[dst], 1);
        csr_src[pos] = src;
    }
}

// ---------------------------------------------------------------- conversions
__global__ void k_f32_to_f16(const float* __restrict__ in, _Float16* __restrict__ out, int n) {
    int i = blockIdx.x * blockDim.x + threadIdx.x;
    if (i < n) out[i] = (_Float16)in[i];
}

// Wt[n][k] = (f16) W[k][n]    (W is K x N row-major, 256x256)
__global__ void k_wt(const float* __restrict__ W, _Float16* __restrict__ Wt) {
    int k = blockIdx.x;      // 0..255
    int n = threadIdx.x;     // 0..255
    Wt[n * DK + k] = (_Float16)W[k * DK + n];
}

// ---------------------------------------------------------------- WMMA GEMM
// C[M x 256] = A[M x 256] (f16, row-major) @ Wt^T (Wt is N-major f16).
// Block = 8 waves; all waves share one 64-col N-block whose B tile (32 KB)
// is staged in LDS once. Each wave owns a 16(M) x 64(N) block: 4 f32
// accumulators, K-loop of 8 steps, A fragment double-buffered from global.
__global__ __launch_bounds__(256) void k_gemm(const _Float16* __restrict__ A,
                                              const _Float16* __restrict__ B,  // Wt, N x K
                                              float* __restrict__ C, int M) {
    __shared__ _Float16 Bs[64 * DK];  // 32 KB
    const int tid = threadIdx.x;
    const int n0 = blockIdx.y * 64;

    // cooperative copy: Wt rows [n0, n0+64) are one contiguous 32 KB span
    {
        const uint4* src = (const uint4*)(B + (size_t)n0 * DK);
        uint4* dst = (uint4*)Bs;
        #pragma unroll
        for (int j = 0; j < 8; ++j)            // 2048 uint4 / 256 threads
            dst[tid + j * 256] = src[tid + j * 256];
    }
    __syncthreads();

    const int lane = tid & 31;
    const int wave = tid >> 5;
    const int m0 = blockIdx.x * 128 + wave * 16;
    if (m0 >= M) return;                        // no barriers after this point
    const int half = lane >> 4;                 // 0: lanes 0-15, 1: lanes 16-31
    const int l16  = lane & 15;
    const int akb  = half * 8;                  // A k sub-offset per ISA layout
    const int bkb  = half * 8;                  // B k sub-offset per ISA layout

    v8f acc[4] = {v8f{}, v8f{}, v8f{}, v8f{}};
    const _Float16* Arow = A + (size_t)(m0 + l16) * DK + akb;

    // software-pipelined A fragment (global), B fragments from LDS
    v8h alo = *(const v8h*)(Arow);
    v8h ahi = *(const v8h*)(Arow + 16);
    #pragma unroll
    for (int k0 = 0; k0 < DK; k0 += 32) {
        v8h nlo = alo, nhi = ahi;
        if (k0 + 32 < DK) {                     // prefetch next k-step's A
            nlo = *(const v8h*)(Arow + k0 + 32);
            nhi = *(const v8h*)(Arow + k0 + 48);
        }
        v16h a = __builtin_shufflevector(alo, ahi, 0, 1, 2, 3, 4, 5, 6, 7,
                                                   8, 9, 10, 11, 12, 13, 14, 15);
        #pragma unroll
        for (int nt = 0; nt < 4; ++nt) {
            const _Float16* Bl = Bs + (nt * 16 + l16) * DK + k0 + bkb;
            v8h blo = *(const v8h*)(Bl);
            v8h bhi = *(const v8h*)(Bl + 16);
            v16h b = __builtin_shufflevector(blo, bhi, 0, 1, 2, 3, 4, 5, 6, 7,
                                                       8, 9, 10, 11, 12, 13, 14, 15);
            acc[nt] = __builtin_amdgcn_wmma_f32_16x16x32_f16(
                false, a, false, b, (short)0, acc[nt], false, false);
        }
        alo = nlo; ahi = nhi;
    }

    // C layout: VGPR v holds row m0 + v + half*8, col = n0 + nt*16 + l16
    const int mb = m0 + half * 8;
    #pragma unroll
    for (int nt = 0; nt < 4; ++nt) {
        const int col = n0 + nt * 16 + l16;
        #pragma unroll
        for (int v = 0; v < 8; ++v)
            C[(size_t)(mb + v) * DK + col] = acc[nt][v];
    }
}

// ---------------------------------------------------------------- aggregation
// out[i] = relu( dinv[i] * ( sum_{s in CSR(i)} dinv[s]*lin[s] + dinv[i]*lin[i] ) + bias )
// one wave per node; lanes cooperatively load 32 edges at a time and broadcast
// via shfl; lane owns 8 contiguous columns; writes f16 (next GEMM's A operand)
__global__ __launch_bounds__(256) void k_agg(const float* __restrict__ lin,
                                             const int* __restrict__ csr_src,
                                             const int* __restrict__ rowptr,
                                             const float* __restrict__ dinv,
                                             const float* __restrict__ bias,
                                             _Float16* __restrict__ out) {
    const int lane = threadIdx.x & 31;
    const int wave = threadIdx.x >> 5;
    const int i = blockIdx.x * 8 + wave;
    if (i >= N_NODES) return;
    const int c0 = lane * 8;

    float acc[8] = {0, 0, 0, 0, 0, 0, 0, 0};
    const int beg = rowptr[i], end = rowptr[i + 1];
    for (int base = beg; base < end; base += 32) {
        const int rem = end - base;
        int   myS = 0;
        float myW = 0.0f;
        if (lane < rem) { myS = csr_src[base + lane]; myW = dinv[myS]; }
        const int cnt = rem < 32 ? rem : 32;
        for (int j = 0; j < cnt; ++j) {
            const int   s = __shfl(myS, j);
            const float w = __shfl(myW, j);
            const float4* p = (const float4*)(lin + (size_t)s * DK + c0);
            float4 x0 = p[0], x1 = p[1];
            acc[0] += w * x0.x; acc[1] += w * x0.y; acc[2] += w * x0.z; acc[3] += w * x0.w;
            acc[4] += w * x1.x; acc[5] += w * x1.y; acc[6] += w * x1.z; acc[7] += w * x1.w;
        }
    }
    const float wi = dinv[i];
    {   // self loop
        const float4* p = (const float4*)(lin + (size_t)i * DK + c0);
        float4 x0 = p[0], x1 = p[1];
        acc[0] += wi * x0.x; acc[1] += wi * x0.y; acc[2] += wi * x0.z; acc[3] += wi * x0.w;
        acc[4] += wi * x1.x; acc[5] += wi * x1.y; acc[6] += wi * x1.z; acc[7] += wi * x1.w;
    }
    v8h h;
    #pragma unroll
    for (int j = 0; j < 8; ++j) {
        float v = acc[j] * wi + bias[c0 + j];
        h[j] = (_Float16)fmaxf(v, 0.0f);
    }
    *(v8h*)(out + (size_t)i * DK + c0) = h;
}

// ---------------------------------------------------------------- pooling + FC
// batch is sorted: accumulate a contiguous 128-node chunk in registers and
// flush to global atomics only on graph-id change (~2 flushes per chunk).
#define POOL_CHUNK 128
__global__ __launch_bounds__(256) void k_pool(const _Float16* __restrict__ h,
                                              const int* __restrict__ batch,
                                              float* __restrict__ pool,
                                              float* __restrict__ cnt) {
    const int d  = threadIdx.x;                 // feature 0..255
    const int i0 = blockIdx.x * POOL_CHUNK;
    const int i1 = (i0 + POOL_CHUNK < N_NODES) ? i0 + POOL_CHUNK : N_NODES;
    int   g   = batch[i0];
    float acc = 0.0f, c = 0.0f;
    for (int i = i0; i < i1; ++i) {
        const int gi = batch[i];
        if (gi != g) {
            atomAddF(&pool[g * DK + d], acc);
            if (d == 0) atomAddF(&cnt[g], c);
            acc = 0.0f; c = 0.0f; g = gi;
        }
        acc += (float)h[(size_t)i * DK + d];
        c   += 1.0f;
    }
    atomAddF(&pool[g * DK + d], acc);
    if (d == 0) atomAddF(&cnt[g], c);
}

__global__ void k_fc(const float* __restrict__ pool, const float* __restrict__ cnt,
                     const float* __restrict__ Wfc, const float* __restrict__ bfc,
                     float* __restrict__ out) {
    int t = blockIdx.x * blockDim.x + threadIdx.x;
    if (t >= N_GRAPHS * D_OUT) return;
    const int g = t >> 4, o = t & 15;
    const float c = fmaxf(cnt[g], 1.0f);
    float s = 0.0f;
    for (int k = 0; k < DK; ++k) s += pool[g * DK + k] * Wfc[k * D_OUT + o];
    out[t] = s / c + bfc[o];
}

// ---------------------------------------------------------------- host launch
extern "C" void kernel_launch(void* const* d_in, const int* in_sizes, int n_in,
                              void* d_out, int out_size, void* d_ws, size_t ws_size,
                              hipStream_t stream) {
    const float* x     = (const float*)d_in[0];
    const int*   ei    = (const int*)d_in[1];
    const int*   batch = (const int*)d_in[2];
    const float* W1 = (const float*)d_in[3];  const float* b1 = (const float*)d_in[4];
    const float* W2 = (const float*)d_in[5];  const float* b2 = (const float*)d_in[6];
    const float* W3 = (const float*)d_in[7];  const float* b3 = (const float*)d_in[8];
    const float* Wfc = (const float*)d_in[9]; const float* bfc = (const float*)d_in[10];
    float* out = (float*)d_out;

    char* ws = (char*)d_ws;
    size_t off = 0;
    auto alloc = [&](size_t bytes) -> void* {
        void* p = ws + off;
        off += (bytes + 255) & ~(size_t)255;
        return p;
    };
    int*      deg    = (int*)alloc(N_NODES * sizeof(int));
    float*    dinv   = (float*)alloc(N_NODES * sizeof(float));
    int*      rowptr = (int*)alloc((N_NODES + 1) * sizeof(int));
    int*      cursor = (int*)alloc(N_NODES * sizeof(int));
    int*      csr    = (int*)alloc(N_EDGES * sizeof(int));
    _Float16* bufA   = (_Float16*)alloc((size_t)N_NODES * DK * sizeof(_Float16));
    float*    lin    = (float*)alloc((size_t)N_NODES * DK * sizeof(float));
    _Float16* Wt     = (_Float16*)alloc((size_t)DK * DK * sizeof(_Float16));
    float*    pool   = (float*)alloc(N_GRAPHS * DK * sizeof(float));  // cnt follows
    float*    cnt    = (float*)alloc(N_GRAPHS * sizeof(float));
    (void)ws_size; (void)in_sizes; (void)n_in; (void)out_size;

    const int T = 256;
    // graph structure (rebuilt every call -> deterministic work)
    k_zero_i32<<<(N_NODES + T - 1) / T, T, 0, stream>>>(deg, N_NODES);
    k_deg<<<(N_EDGES + T - 1) / T, T, 0, stream>>>(ei, deg);
    k_dinv<<<(N_NODES + T - 1) / T, T, 0, stream>>>(deg, dinv);
    k_scan<<<1, 1024, 0, stream>>>(deg, rowptr, cursor);
    k_fill<<<(N_EDGES + T - 1) / T, T, 0, stream>>>(ei, cursor, csr);

    // first GEMM input
    k_f32_to_f16<<<(N_NODES * DK + T - 1) / T, T, 0, stream>>>(x, bufA, N_NODES * DK);

    const float* Ws[3] = {W1, W2, W3};
    const float* bs[3] = {b1, b2, b3};
    dim3 ggrid((N_NODES + 127) / 128, 4);
    for (int l = 0; l < 3; ++l) {
        k_wt<<<DK, DK, 0, stream>>>(Ws[l], Wt);
        k_gemm<<<ggrid, 256, 0, stream>>>(bufA, Wt, lin, N_NODES);
        k_agg<<<(N_NODES + 7) / 8, 256, 0, stream>>>(lin, csr, rowptr, dinv, bs[l], bufA);
    }

    // pooling (pool and cnt are contiguous: zero both in one pass)
    k_zero_f32<<<(N_GRAPHS * DK + N_GRAPHS + T - 1) / T, T, 0, stream>>>(pool, N_GRAPHS * DK + N_GRAPHS);
    k_pool<<<(N_NODES + POOL_CHUNK - 1) / POOL_CHUNK, DK, 0, stream>>>(bufA, batch, pool, cnt);
    k_fc<<<(N_GRAPHS * D_OUT + T - 1) / T, T, 0, stream>>>(pool, cnt, Wfc, bfc, out);
}
